// ModernBertAttention_34531537059987
// MI455X (gfx1250) — matrix operand order repaired and checked by reading, not compiled, not run on hardware
//
#include <hip/hip_runtime.h>
#include <hip/hip_fp16.h>

typedef __attribute__((ext_vector_type(16))) _Float16 v16h;
typedef __attribute__((ext_vector_type(8)))  float    v8f;

#define D_MODEL 768
#define N_HEADS 12
#define HEAD_DIM 64
#define WINDOW 128

// CDNA5 async global->LDS DMA (ASYNCcnt-tracked). Flip to 0 to revert to the
// register-staged double buffer if the assembler rejects the mnemonics.
#define USE_ASYNC_LDS 1

__device__ __forceinline__ v8f zero8() {
    v8f v;
#pragma unroll
    for (int i = 0; i < 8; ++i) v[i] = 0.0f;
    return v;
}

#if USE_ASYNC_LDS
__device__ __forceinline__ void async_load_b128(const _Float16* g, _Float16* lds)
{
    // addr[31:0] of the shared-aperture flat address is the LDS byte offset
    unsigned           l  = (unsigned)(size_t)lds;
    unsigned long long ga = (unsigned long long)(size_t)g;
    asm volatile("global_load_async_to_lds_b128 %0, %1, off"
                 :: "v"(l), "v"(ga) : "memory");
}
__device__ __forceinline__ void wait_async()
{
    asm volatile("s_wait_asynccnt 0x0" ::: "memory");
}
#endif

// ---------------------------------------------------------------------------
// RoPE (rotate-half, theta=1e4, applied to raw hidden) + f16 pack of both the
// RoPE'd stream (Q/K input) and the raw stream (V input).
// ---------------------------------------------------------------------------
__global__ void rope_pack_kernel(const float* __restrict__ hs,
                                 _Float16* __restrict__ qk16,
                                 _Float16* __restrict__ h16,
                                 int total, int S)
{
    int idx = blockIdx.x * blockDim.x + threadIdx.x;
    if (idx >= total) return;
    int d   = idx % D_MODEL;
    int bs  = idx / D_MODEL;
    int pos = bs % S;
    int dh  = d & (HEAD_DIM - 1);
    int hf  = dh & 31;
    float expo = (float)(2 * hf) * (1.0f / 64.0f);
    float invf = __powf(10000.0f, -expo);
    float ang  = (float)pos * invf;
    float sn, cs;
    __sincosf(ang, &sn, &cs);
    float x   = hs[idx];
    float rot = (dh < 32) ? -hs[idx + 32] : hs[idx - 32];
    qk16[idx] = (_Float16)(x * cs + rot * sn);
    h16[idx]  = (_Float16)x;
}

// Pack weight f32 [K][N] -> f16 transposed [N][K] (coalesced on output).
__global__ void pack_weight_T_kernel(const float* __restrict__ W,
                                     _Float16* __restrict__ WT, int Kd, int Nd)
{
    int i = blockIdx.x * blockDim.x + threadIdx.x;
    if (i >= Kd * Nd) return;
    int n = i / Kd;
    int k = i % Kd;
    WT[i] = (_Float16)W[(size_t)k * Nd + n];
}

// ---------------------------------------------------------------------------
// Double-buffered WMMA GEMM: C[M,N] = (A[M,K] * B[K,N] + bias[N]) * scale
// A row-major f16; weights pre-transposed BT[N][K] so both LDS tiles stage
// with b128 transfers. 8 waves, 128x64 block tile, 32x32 per wave.
// Tiles for step kt+1 are DMA'd (async global->LDS) into the spare buffer at
// the top of step kt; one s_wait_asynccnt + one barrier per step.
// storeMode: 0 = f32 row-major, 1 = f16 row-major,
//            2 = f16 col-major out[(b*N+col)*S2 + s]  (b=row/S2, s=row%S2)
// ---------------------------------------------------------------------------
__global__ void __launch_bounds__(256)
gemm_f16_wmma(const _Float16* __restrict__ A, const _Float16* __restrict__ BT,
              const float* __restrict__ bias, void* __restrict__ Cout,
              int M, int N, int K, float scale, int storeMode, int S2)
{
    __shared__ __align__(16) _Float16 As[2][128][40];
    __shared__ __align__(16) _Float16 Bs[2][64][40];

    const int tid  = threadIdx.x;
    const int lane = tid & 31;
    const int wave = tid >> 5;
    const int wm = (wave & 3) * 32;
    const int wn = (wave >> 2) * 32;
    const int blockM = blockIdx.y * 128;
    const int blockN = blockIdx.x * 64;

    const int arow = tid >> 1, akoff = (tid & 1) * 16;
    const int bcol = tid >> 2, bkoff = (tid & 3) * 8;
    const _Float16* aSrc = A  + (size_t)(blockM + arow) * K + akoff;
    const _Float16* bSrc = BT + (size_t)(blockN + bcol) * K + bkoff;

    v8f acc[2][2];
#pragma unroll
    for (int mi = 0; mi < 2; ++mi)
#pragma unroll
        for (int ni = 0; ni < 2; ++ni) acc[mi][ni] = zero8();

    const int nk = K / 32;

#if USE_ASYNC_LDS
    // preload tile 0 via async DMA
    async_load_b128(aSrc,     &As[0][arow][akoff]);
    async_load_b128(aSrc + 8, &As[0][arow][akoff + 8]);
    async_load_b128(bSrc,     &Bs[0][bcol][bkoff]);
    wait_async();
#else
    {
        uint4 ra0 = *(const uint4*)aSrc;
        uint4 ra1 = *(const uint4*)(aSrc + 8);
        uint4 rb0 = *(const uint4*)bSrc;
        *(uint4*)&As[0][arow][akoff]     = ra0;
        *(uint4*)&As[0][arow][akoff + 8] = ra1;
        *(uint4*)&Bs[0][bcol][bkoff]     = rb0;
    }
#endif
    __syncthreads();

    const int ka = (lane >> 4) * 8;
    const int kb = (lane >> 4) * 8;

    for (int kt = 0; kt < nk; ++kt) {
        const int  cur     = kt & 1;
        const bool hasNext = (kt + 1 < nk);
#if USE_ASYNC_LDS
        if (hasNext) {
            // safe: barrier at end of step kt-1 guarantees nobody still reads
            // buffer cur^1; DMA overlaps with this step's WMMAs
            const _Float16* an = aSrc + (kt + 1) * 32;
            const _Float16* bn = bSrc + (kt + 1) * 32;
            async_load_b128(an,     &As[cur ^ 1][arow][akoff]);
            async_load_b128(an + 8, &As[cur ^ 1][arow][akoff + 8]);
            async_load_b128(bn,     &Bs[cur ^ 1][bcol][bkoff]);
            if (kt + 2 < nk) {
                __builtin_prefetch(aSrc + (kt + 2) * 32, 0, 1);
                __builtin_prefetch(bSrc + (kt + 2) * 32, 0, 1);
            }
        }
#else
        uint4 na0, na1, nb0;
        if (hasNext) {
            const _Float16* an = aSrc + (kt + 1) * 32;
            const _Float16* bn = bSrc + (kt + 1) * 32;
            na0 = *(const uint4*)an;
            na1 = *(const uint4*)(an + 8);
            nb0 = *(const uint4*)bn;
            if (kt + 2 < nk) {
                __builtin_prefetch(aSrc + (kt + 2) * 32, 0, 1);
                __builtin_prefetch(bSrc + (kt + 2) * 32, 0, 1);
            }
        }
#endif

        v16h af[2], bf[2];
#pragma unroll
        for (int mi = 0; mi < 2; ++mi) {
            int row = wm + mi * 16 + (lane & 15);
            ((uint4*)&af[mi])[0] = *(const uint4*)&As[cur][row][ka];
            ((uint4*)&af[mi])[1] = *(const uint4*)&As[cur][row][ka + 16];
        }
#pragma unroll
        for (int ni = 0; ni < 2; ++ni) {
            int col = wn + ni * 16 + (lane & 15);
            ((uint4*)&bf[ni])[0] = *(const uint4*)&Bs[cur][col][kb];
            ((uint4*)&bf[ni])[1] = *(const uint4*)&Bs[cur][col][kb + 16];
        }
#pragma unroll
        for (int mi = 0; mi < 2; ++mi)
#pragma unroll
            for (int ni = 0; ni < 2; ++ni)
                acc[mi][ni] = __builtin_amdgcn_wmma_f32_16x16x32_f16(
                    false, af[mi], false, bf[ni], (short)0, acc[mi][ni], false, false);

        if (hasNext) {
#if USE_ASYNC_LDS
            wait_async();      // my DMA slice landed
            __syncthreads();   // everyone's slice landed & everyone done reading
#else
            __syncthreads();
            *(uint4*)&As[cur ^ 1][arow][akoff]     = na0;
            *(uint4*)&As[cur ^ 1][arow][akoff + 8] = na1;
            *(uint4*)&Bs[cur ^ 1][bcol][bkoff]     = nb0;
            __syncthreads();
#endif
        }
    }

    // epilogue: C lane layout M = r + 8*(lane>>4), N = lane&15
#pragma unroll
    for (int mi = 0; mi < 2; ++mi)
#pragma unroll
        for (int ni = 0; ni < 2; ++ni) {
            const int col  = blockN + wn + ni * 16 + (lane & 15);
            const float bb = bias ? bias[col] : 0.0f;
            const int row0 = blockM + wm + mi * 16 + (lane >> 4) * 8;
            if (storeMode == 2) {
                union { uint4 u; _Float16 h[8]; } tv;
#pragma unroll
                for (int r = 0; r < 8; ++r)
                    tv.h[r] = (_Float16)((acc[mi][ni][r] + bb) * scale);
                int bIdx = row0 / S2, s0 = row0 % S2;
                _Float16* dst = (_Float16*)Cout + ((size_t)bIdx * N + col) * S2 + s0;
                *(uint4*)dst = tv.u;
            } else if (storeMode == 1) {
#pragma unroll
                for (int r = 0; r < 8; ++r)
                    ((_Float16*)Cout)[(size_t)(row0 + r) * N + col] =
                        (_Float16)((acc[mi][ni][r] + bb) * scale);
            } else {
#pragma unroll
                for (int r = 0; r < 8; ++r)
                    ((float*)Cout)[(size_t)(row0 + r) * N + col] =
                        (acc[mi][ni][r] + bb) * scale;
            }
        }
}

// ---------------------------------------------------------------------------
// Windowed flash attention, 1 wave per (b, h, 16-query tile). No LDS, no
// barriers. Scores computed transposed: S^T = K * Q^T, so the S^T accumulator
// layout IS the A-fragment layout for P in the P*V WMMA (pair of 16-key
// tiles -> K=32), and V is pre-transposed in global ([b*768+col][key]).
// Q pre-scaled by 1/sqrt(Dh) in its GEMM epilogue.
// ---------------------------------------------------------------------------
__global__ void __launch_bounds__(32)
attn_wmma(const _Float16* __restrict__ Qh, const _Float16* __restrict__ Kh,
          const _Float16* __restrict__ VtG, _Float16* __restrict__ Ctx, int S)
{
    const int lane = threadIdx.x;
    const int nqt  = S / 16;
    const int id   = blockIdx.x;
    const int q0   = (id % nqt) * 16;
    const int h    = (id / nqt) % N_HEADS;
    const int b    = id / (nqt * N_HEADS);

    const int lq = lane & 15;      // query column owned by this lane (S^T layout)
    const int hi = lane >> 4;
    const int ka = hi * 8;
    const int kb = hi * 8;

    // Q^T B-fragments: lane = col q, halves = dh (contiguous row reads)
    v16h qtf[2];
    {
        const _Float16* qp = Qh + ((size_t)b * S + q0 + lq) * D_MODEL + h * HEAD_DIM;
#pragma unroll
        for (int c = 0; c < 2; ++c) {
            ((uint4*)&qtf[c])[0] = *(const uint4*)(qp + c * 32 + kb);
            ((uint4*)&qtf[c])[1] = *(const uint4*)(qp + c * 32 + kb + 16);
        }
    }

    float mstate = -1e30f, lstate = 0.0f;   // per q = lq (dup in lane lq+16)
    v8f fc[4];
#pragma unroll
    for (int t = 0; t < 4; ++t) fc[t] = zero8();

    const _Float16* vtb = VtG + (size_t)(b * D_MODEL + h * HEAD_DIM) * S;

    int lo  = q0 - (WINDOW - 1); if (lo < 0) lo = 0;
    int hiK = q0 + 15 + (WINDOW - 1); if (hiK > S - 1) hiK = S - 1;
    const int kt_lo = lo >> 4, kt_hi = hiK >> 4;

    for (int kt = kt_lo; kt <= kt_hi; kt += 2) {
        const int nvalid = (kt + 1 <= kt_hi) ? 2 : 1;

        // ---- scores S^T = K * Q^T (lane: q = lq; element r: key = tile+r+8*hi)
        v8f sc[2];
#pragma unroll
        for (int i = 0; i < 8; ++i) sc[1][i] = -1e30f;  // phantom tile -> p=0
        for (int t = 0; t < nvalid; ++t) {
            const int keyrow = (kt + t) * 16 + lq;      // A row = key
            const _Float16* kp = Kh + ((size_t)b * S + keyrow) * D_MODEL + h * HEAD_DIM;
            v8f s = zero8();
#pragma unroll
            for (int c = 0; c < 2; ++c) {
                v16h kf;
                ((uint4*)&kf)[0] = *(const uint4*)(kp + c * 32 + ka);
                ((uint4*)&kf)[1] = *(const uint4*)(kp + c * 32 + 16 + ka);
                s = __builtin_amdgcn_wmma_f32_16x16x32_f16(
                        false, kf, false, qtf[c], (short)0, s, false, false);
            }
            const int q = q0 + lq;
#pragma unroll
            for (int r = 0; r < 8; ++r) {
                int key = (kt + t) * 16 + r + 8 * hi;
                int dd = q - key; if (dd < 0) dd = -dd;
                if (dd >= WINDOW) s[r] += -1e9f;
            }
            sc[t] = s;
        }

        // ---- online softmax (in-lane over 8 keys x 2 tiles + one xor-16)
        float tm = sc[0][0];
#pragma unroll
        for (int r = 1; r < 8; ++r) tm = fmaxf(tm, sc[0][r]);
#pragma unroll
        for (int r = 0; r < 8; ++r) tm = fmaxf(tm, sc[1][r]);
        tm = fmaxf(tm, __shfl_xor(tm, 16, 32));
        const float mnew  = fmaxf(mstate, tm);
        const float alpha = __expf(mstate - mnew);
        mstate = mnew;

        union { v16h v; _Float16 hh[16]; } pu;
        float ps = 0.0f;
#pragma unroll
        for (int t = 0; t < 2; ++t)
#pragma unroll
            for (int r = 0; r < 8; ++r) {
                float p = __expf(sc[t][r] - mnew);
                ps += p;
                pu.hh[t * 8 + r] = (_Float16)p;   // == A-frag halves, in order
            }
        ps += __shfl_xor(ps, 16, 32);
        lstate = lstate * alpha + ps;

        // rescale ctx rows: lane's fc element r is q = r + 8*hi
#pragma unroll
        for (int r = 0; r < 8; ++r) {
            float av = __shfl(alpha, r + 8 * hi, 32);
#pragma unroll
            for (int t = 0; t < 4; ++t) fc[t][r] *= av;
        }

        // ---- ctx += P(16q x 32key) * V(32key x 64dh): 4 WMMAs ----
#pragma unroll
        for (int t = 0; t < 4; ++t) {
            const _Float16* vp = vtb + (size_t)(t * 16 + lq) * S + kt * 16 + kb;
            v16h vf;
            ((uint4*)&vf)[0] = *(const uint4*)vp;
            ((uint4*)&vf)[1] = *(const uint4*)(vp + 16);
            fc[t] = __builtin_amdgcn_wmma_f32_16x16x32_f16(
                        false, pu.v, false, vf, (short)0, fc[t], false, false);
        }
        if (kt + 2 <= kt_hi)
            __builtin_prefetch(Kh + ((size_t)b * S + (kt + 2) * 16 + lq) * D_MODEL + h * HEAD_DIM, 0, 1);
    }

    // ---- normalize (broadcast 1/l per output row) and store ctx ----
    const float linv = 1.0f / lstate;
#pragma unroll
    for (int r = 0; r < 8; ++r) {
        const float lr = __shfl(linv, r + 8 * hi, 32);
        const int   qr = q0 + r + 8 * hi;
        _Float16* op = Ctx + ((size_t)b * S + qr) * D_MODEL + h * HEAD_DIM;
#pragma unroll
        for (int t = 0; t < 4; ++t)
            op[t * 16 + lq] = (_Float16)(fc[t][r] * lr);
    }
}

// ---------------------------------------------------------------------------
extern "C" void kernel_launch(void* const* d_in, const int* in_sizes, int n_in,
                              void* d_out, int out_size, void* d_ws, size_t ws_size,
                              hipStream_t stream)
{
    const float* hs = (const float*)d_in[0];
    const float* Wq = (const float*)d_in[1];
    const float* bq = (const float*)d_in[2];
    const float* Wk = (const float*)d_in[3];
    const float* bk = (const float*)d_in[4];
    const float* Wv = (const float*)d_in[5];
    const float* bv = (const float*)d_in[6];
    const float* Wo = (const float*)d_in[7];
    const float* bo = (const float*)d_in[8];
    float* out = (float*)d_out;

    const int S  = 2048;
    const int BS = in_sizes[0] / D_MODEL;     // B*S = 4096
    const int Bb = BS / S;
    const int total  = BS * D_MODEL;
    const int wElems = D_MODEL * D_MODEL;

    const size_t actB = (size_t)BS * D_MODEL * sizeof(_Float16);
    const size_t wB   = (size_t)wElems * sizeof(_Float16);

    char* p = (char*)d_ws;
    _Float16* qk16 = (_Float16*)p; p += actB;
    _Float16* h16  = (_Float16*)p; p += actB;
    _Float16* wqT  = (_Float16*)p; p += wB;
    _Float16* wkT  = (_Float16*)p; p += wB;
    _Float16* wvT  = (_Float16*)p; p += wB;
    _Float16* woT  = (_Float16*)p; p += wB;
    _Float16* Qh   = (_Float16*)p; p += actB;
    _Float16* Kp   = (_Float16*)p; p += actB;
    _Float16* VtG  = (_Float16*)p; p += actB;   // followed by Ctx: tail-read safe
    _Float16* Ctx  = (_Float16*)p; p += actB;

    // 1) RoPE + pack
    rope_pack_kernel<<<(total + 255) / 256, 256, 0, stream>>>(hs, qk16, h16, total, S);

    // 2) weight packs (f32 [K][N] -> f16 [N][K])
    pack_weight_T_kernel<<<(wElems + 255) / 256, 256, 0, stream>>>(Wq, wqT, D_MODEL, D_MODEL);
    pack_weight_T_kernel<<<(wElems + 255) / 256, 256, 0, stream>>>(Wk, wkT, D_MODEL, D_MODEL);
    pack_weight_T_kernel<<<(wElems + 255) / 256, 256, 0, stream>>>(Wv, wvT, D_MODEL, D_MODEL);
    pack_weight_T_kernel<<<(wElems + 255) / 256, 256, 0, stream>>>(Wo, woT, D_MODEL, D_MODEL);

    // 3) projections (Q folds the 1/sqrt(64) scale; V stored transposed)
    dim3 ggrid(D_MODEL / 64, BS / 128);
    gemm_f16_wmma<<<ggrid, 256, 0, stream>>>(qk16, wqT, bq, Qh,  BS, D_MODEL, D_MODEL, 0.125f, 1, S);
    gemm_f16_wmma<<<ggrid, 256, 0, stream>>>(qk16, wkT, bk, Kp,  BS, D_MODEL, D_MODEL, 1.0f,   1, S);
    gemm_f16_wmma<<<ggrid, 256, 0, stream>>>(h16,  wvT, bv, VtG, BS, D_MODEL, D_MODEL, 1.0f,   2, S);

    // 4) windowed flash attention (LDS-free, barrier-free)
    const int nblk = Bb * N_HEADS * (S / 16);
    attn_wmma<<<nblk, 32, 0, stream>>>(Qh, Kp, VtG, Ctx, S);

    // 5) output projection -> f32
    gemm_f16_wmma<<<ggrid, 256, 0, stream>>>(Ctx, woT, bo, out, BS, D_MODEL, D_MODEL, 1.0f, 0, S);
}
